// MultiHeadAttention_63943473103080
// MI455X (gfx1250) — matrix operand order repaired and checked by reading, not compiled, not run on hardware
//
#include <hip/hip_runtime.h>
#include <hip/hip_bf16.h>
#include <stdint.h>

#ifndef __has_builtin
#define __has_builtin(x) 0
#endif

#if __has_builtin(__builtin_amdgcn_global_load_async_to_lds_b128)
#define USE_ASYNC 1
#else
#define USE_ASYNC 0
#endif

typedef __attribute__((ext_vector_type(16))) __bf16 v16bf;
typedef __attribute__((ext_vector_type(8)))  float  v8f;
typedef int v4i __attribute__((vector_size(16)));   // matches builtin param type

constexpr int Bc = 4, Sc = 2048, Ec = 1024, Hc = 16, Dc = 64;
constexpr int LSTR = 72;          // padded LDS row stride (bf16 elems) = 144 B
constexpr float QSCALE = 0.125f;  // 1/sqrt(64)

__device__ __forceinline__ unsigned short f2bf(float f) {
  unsigned u = __float_as_uint(f);
  u += 0x7FFFu + ((u >> 16) & 1u);          // round-to-nearest-even
  return (unsigned short)(u >> 16);
}
__device__ __forceinline__ __bf16 us2bf(unsigned short s) {
  union { unsigned short u; __bf16 b; } x; x.u = s; return x.b;
}
__device__ __forceinline__ void lds_fence() {
  asm volatile("s_wait_dscnt 0x0" ::: "memory");
}
__device__ __forceinline__ void async_wait() {
#if USE_ASYNC
#if __has_builtin(__builtin_amdgcn_s_wait_asynccnt)
  __builtin_amdgcn_s_wait_asynccnt(0);
#else
  asm volatile("s_wait_asynccnt 0x0" ::: "memory");
#endif
#endif
}

// Copy one 64x64 bf16 tile (row-major, contiguous 128B rows in global) into
// padded LDS [64][LSTR]; 128 threads, 4 x 16B segments each.
__device__ __forceinline__ void copy_tile_g2l(unsigned short* lds,
                                              const unsigned short* g,
                                              int tid) {
#if USE_ASYNC
#pragma unroll
  for (int i = 0; i < 4; ++i) {
    int idx = tid * 4 + i;
    int row = idx >> 3, seg = idx & 7;
    const unsigned short* gp = g + row * 64 + seg * 8;
    unsigned short* lp = lds + row * LSTR + seg * 8;
    __builtin_amdgcn_global_load_async_to_lds_b128(
        (v4i*)gp, (v4i*)lp, 0, 0);
  }
#else
#pragma unroll
  for (int i = 0; i < 4; ++i) {
    int idx = tid * 4 + i;
    int row = idx >> 3, seg = idx & 7;
    uint4 v = *(const uint4*)(g + row * 64 + seg * 8);
    *(uint4*)(lds + row * LSTR + seg * 8) = v;
  }
#endif
}

// A fragment (16x32 bf16, MxK). m = lane%16 (+rowBase). ISA layout:
// VGPR j<4: k = half*8 + 2j ; VGPR j>=4: k = 16 + half*8 + 2(j-4).
__device__ __forceinline__ v16bf load_a(const unsigned short* lds, int lane,
                                        int rowBase, int kBase) {
  int half = lane >> 4, m = lane & 15;
  const unsigned short* r = lds + (rowBase + m) * LSTR + kBase;
  v16bf a;
#pragma unroll
  for (int j = 0; j < 8; ++j) {
    int kk = (j < 4) ? (half * 8 + 2 * j) : (16 + half * 8 + 2 * (j - 4));
    a[2 * j]     = us2bf(r[kk]);
    a[2 * j + 1] = us2bf(r[kk + 1]);
  }
  return a;
}

// B fragment (32x16 bf16, KxN), element (k,n) = lds[(nBase+n)*LSTR + kBase + k].
// ISA layout: VGPR j<4: k = half*8 + 2j ; VGPR j>=4: k = 16 + half*8 + 2(j-4).
__device__ __forceinline__ v16bf load_b(const unsigned short* lds, int lane,
                                        int nBase, int kBase) {
  int half = lane >> 4, n = lane & 15;
  const unsigned short* r = lds + (nBase + n) * LSTR + kBase;
  v16bf b;
#pragma unroll
  for (int j = 0; j < 8; ++j) {
    int kk = (j < 4) ? (half * 8 + 2 * j) : (16 + half * 8 + 2 * (j - 4));
    b[2 * j]     = us2bf(r[kk]);
    b[2 * j + 1] = us2bf(r[kk + 1]);
  }
  return b;
}

__device__ __forceinline__ v8f wmma_bf16(v16bf a, v16bf b, v8f c) {
  return __builtin_amdgcn_wmma_f32_16x16x32_bf16(false, a, false, b,
                                                 (short)0, c, false, false);
}

// ---------------- Kernel 1: fused QKV projection (bf16 out) ----------------
__global__ __launch_bounds__(128) void qkv_kernel(
    const float* __restrict__ x,
    const float* __restrict__ Wq, const float* __restrict__ bq,
    const float* __restrict__ Wk, const float* __restrict__ bk,
    const float* __restrict__ Wv, const float* __restrict__ bv,
    unsigned short* __restrict__ Qb, unsigned short* __restrict__ Kb,
    unsigned short* __restrict__ Vb) {
  __shared__ alignas(16) unsigned short Xs[64 * LSTR];
  __shared__ alignas(16) unsigned short Wt[3][64 * LSTR];

  const int tid = threadIdx.x;
  const int lane = tid & 31, wave = tid >> 5;
  const int tilesPerBH = Sc / 64;
  const int bh = blockIdx.x / tilesPerBH;
  const int b = bh / Hc, h = bh % Hc;
  const int s0 = (blockIdx.x % tilesPerBH) * 64;

  {  // x tile [64 tokens x 64 dims] -> bf16 LDS
    int r = tid & 63, hs = tid >> 6;
    const float* src = x + ((size_t)b * Sc + (s0 + r)) * Ec + h * Dc + hs * 32;
    unsigned short* dst = Xs + r * LSTR + hs * 32;
#pragma unroll
    for (int i = 0; i < 32; ++i) dst[i] = f2bf(src[i]);
  }
  const float* Wsrc[3] = { Wq + (size_t)h * Dc * Dc, Wk + (size_t)h * Dc * Dc,
                           Wv + (size_t)h * Dc * Dc };
#pragma unroll
  for (int p = 0; p < 3; ++p) {  // W transposed: Wt[e][d] = W[d][e]
    int r = tid & 63, hs = tid >> 6;
    const float* src = Wsrc[p] + r * Dc + hs * 32;
#pragma unroll
    for (int i = 0; i < 32; ++i) Wt[p][(hs * 32 + i) * LSTR + r] = f2bf(src[i]);
  }
  __syncthreads();

  v16bf a0 = load_a(Xs, lane, wave * 16, 0);
  v16bf a1 = load_a(Xs, lane, wave * 16, 32);

  const float* bias[3] = { bq + h * Dc, bk + h * Dc, bv + h * Dc };
  unsigned short* outp[3] = { Qb, Kb, Vb };
  const int n0 = lane & 15, hl = lane >> 4;
  const size_t base = ((size_t)bh * Sc + s0 + wave * 16) * Dc;

#pragma unroll
  for (int p = 0; p < 3; ++p) {
#pragma unroll
    for (int nt = 0; nt < 4; ++nt) {
      float bval = bias[p][nt * 16 + n0];
      v8f c;
#pragma unroll
      for (int r = 0; r < 8; ++r) c[r] = bval;
      c = wmma_bf16(a0, load_b(Wt[p], lane, nt * 16, 0), c);
      c = wmma_bf16(a1, load_b(Wt[p], lane, nt * 16, 32), c);
      float sc = (p == 0) ? QSCALE : 1.0f;
#pragma unroll
      for (int r = 0; r < 8; ++r)
        outp[p][base + (size_t)(r + 8 * hl) * Dc + nt * 16 + n0] = f2bf(c[r] * sc);
    }
  }
}

// ---------------- Kernel 2: flash attention ----------------
__global__ __launch_bounds__(128) void attn_kernel(
    const unsigned short* __restrict__ Qb,
    const unsigned short* __restrict__ Kb,
    const unsigned short* __restrict__ Vb,
    float* __restrict__ out) {
  __shared__ alignas(16) unsigned short Qs[64 * LSTR];
  __shared__ alignas(16) unsigned short Kc[64 * LSTR];
  __shared__ alignas(16) unsigned short Vt[64 * LSTR];   // transposed: Vt[d][key]
  __shared__ alignas(16) unsigned short Pb[4][16 * LSTR];

  const int tid = threadIdx.x, lane = tid & 31, wave = tid >> 5;
  const int tilesPerBH = Sc / 64;
  const int bh = blockIdx.x / tilesPerBH;
  const int b = bh / Hc, h = bh % Hc;
  const int q0 = (blockIdx.x % tilesPerBH) * 64;
  const int n0 = lane & 15, hl = lane >> 4;

  copy_tile_g2l(Qs, Qb + ((size_t)bh * Sc + q0) * Dc, tid);
  async_wait();
  __syncthreads();

  v16bf qa0 = load_a(Qs, lane, wave * 16, 0);
  v16bf qa1 = load_a(Qs, lane, wave * 16, 32);

  v8f o[4];
#pragma unroll
  for (int t = 0; t < 4; ++t)
#pragma unroll
    for (int r = 0; r < 8; ++r) o[t][r] = 0.f;
  float rowmax[8], rowsum[8];
#pragma unroll
  for (int r = 0; r < 8; ++r) { rowmax[r] = -__builtin_inff(); rowsum[r] = 0.f; }

  for (int kc = 0; kc < Sc / 64; ++kc) {
    __syncthreads();  // previous chunk's reads of Kc/Vt complete
    copy_tile_g2l(Kc, Kb + ((size_t)bh * Sc + kc * 64) * Dc, tid);
    {  // V chunk transposed into Vt[d][key]
      int r = tid & 63, hs = tid >> 6;
      const unsigned short* src =
          Vb + ((size_t)bh * Sc + kc * 64 + r) * Dc + hs * 32;
#pragma unroll
      for (int i = 0; i < 32; ++i) Vt[(hs * 32 + i) * LSTR + r] = src[i];
    }
    async_wait();
    __syncthreads();

    // scores S = Q K^T (Q pre-scaled by 1/sqrt(D))
    v8f sc[4];
#pragma unroll
    for (int nt = 0; nt < 4; ++nt) {
      v8f c;
#pragma unroll
      for (int r = 0; r < 8; ++r) c[r] = 0.f;
      c = wmma_bf16(qa0, load_b(Kc, lane, nt * 16, 0), c);
      c = wmma_bf16(qa1, load_b(Kc, lane, nt * 16, 32), c);
      sc[nt] = c;
    }

    // online softmax; row m = r + 8*(lane/16) lives across 16 lanes of a half
#pragma unroll
    for (int r = 0; r < 8; ++r) {
      float m = fmaxf(fmaxf(sc[0][r], sc[1][r]), fmaxf(sc[2][r], sc[3][r]));
#pragma unroll
      for (int off = 8; off >= 1; off >>= 1) m = fmaxf(m, __shfl_xor(m, off, 32));
      float newmax = fmaxf(rowmax[r], m);
      float corr = __expf(rowmax[r] - newmax);
      rowmax[r] = newmax;
      float psum = 0.f;
#pragma unroll
      for (int nt = 0; nt < 4; ++nt) {
        float p = __expf(sc[nt][r] - newmax);
        sc[nt][r] = p;
        psum += p;
      }
#pragma unroll
      for (int off = 8; off >= 1; off >>= 1) psum += __shfl_xor(psum, off, 32);
      rowsum[r] = rowsum[r] * corr + psum;
#pragma unroll
      for (int t = 0; t < 4; ++t) o[t][r] *= corr;
    }

    // P -> per-wave LDS (bf16), reshaped into A-fragment layout for PV
    unsigned short* pw = Pb[wave];
#pragma unroll
    for (int nt = 0; nt < 4; ++nt)
#pragma unroll
      for (int r = 0; r < 8; ++r)
        pw[(r + 8 * hl) * LSTR + nt * 16 + n0] = f2bf(sc[nt][r]);
    lds_fence();

    v16bf pa0 = load_a(pw, lane, 0, 0);
    v16bf pa1 = load_a(pw, lane, 0, 32);
#pragma unroll
    for (int dt = 0; dt < 4; ++dt) {
      o[dt] = wmma_bf16(pa0, load_b(Vt, lane, dt * 16, 0), o[dt]);
      o[dt] = wmma_bf16(pa1, load_b(Vt, lane, dt * 16, 32), o[dt]);
    }
  }

  // out[b, s, h*64 + d] = O / rowsum
#pragma unroll
  for (int dt = 0; dt < 4; ++dt)
#pragma unroll
    for (int r = 0; r < 8; ++r) {
      int srow = q0 + wave * 16 + r + 8 * hl;
      out[((size_t)b * Sc + srow) * Ec + h * Dc + dt * 16 + n0] =
          o[dt][r] / rowsum[r];
    }
}

extern "C" void kernel_launch(void* const* d_in, const int* in_sizes, int n_in,
                              void* d_out, int out_size, void* d_ws,
                              size_t ws_size, hipStream_t stream) {
  const float* x  = (const float*)d_in[0];
  const float* Wq = (const float*)d_in[1];
  const float* bq = (const float*)d_in[2];
  const float* Wk = (const float*)d_in[3];
  const float* bk = (const float*)d_in[4];
  const float* Wv = (const float*)d_in[5];
  const float* bv = (const float*)d_in[6];
  float* out = (float*)d_out;

  size_t per = (size_t)Bc * Hc * Sc * Dc;  // elements per Q/K/V buffer
  unsigned short* Qb = (unsigned short*)d_ws;
  unsigned short* Kb = Qb + per;
  unsigned short* Vb = Kb + per;

  int blocks = Bc * Hc * (Sc / 64);  // 2048
  qkv_kernel<<<blocks, 128, 0, stream>>>(x, Wq, bq, Wk, bk, Wv, bv, Qb, Kb, Vb);
  attn_kernel<<<blocks, 128, 0, stream>>>(Qb, Kb, Vb, out);
}
